// LiquidLayer_12463995093872
// MI455X (gfx1250) — matrix operand order repaired and checked by reading, not compiled, not run on hardware
//
#include <hip/hip_runtime.h>
#include <hip/hip_bf16.h>
#include <math.h>
#include <stdint.h>

// ---------------------------------------------------------------------------
// LiquidCell forward for MI455X (gfx1250, wave32, WMMA + TDM).
//   xW = x @ W^T   (bf16 WMMA GEMM, fp32 accum, written into d_out states)
//   for t: h = h + DT*(-h/tau + tanh(xW_t + h @ U^T + b))   (persistent kernel,
//   U slice staged in LDS once via tensor_load_to_lds)
// ---------------------------------------------------------------------------

typedef __attribute__((ext_vector_type(16))) __bf16 bf16x16;
typedef __attribute__((ext_vector_type(8)))  __bf16 bf16x8;
typedef __attribute__((ext_vector_type(8)))  float  f32x8;
typedef __attribute__((ext_vector_type(4)))  unsigned int u32x4;
typedef __attribute__((ext_vector_type(8)))  int i32x8;
typedef __attribute__((ext_vector_type(4)))  int i32x4;

#define DT_STEP 0.1f
static constexpr int Bdim = 32;   // batch (fixed by reference)

__device__ inline unsigned rfl(unsigned v) {
  return (unsigned)__builtin_amdgcn_readfirstlane((int)v);
}

// ---------------- helpers ----------------

__global__ void cvt_bf16_kernel(const float* __restrict__ s,
                                __bf16* __restrict__ d, int n) {
  int i = blockIdx.x * blockDim.x + threadIdx.x;
  int stride = gridDim.x * blockDim.x;
  for (; i < n; i += stride) d[i] = (__bf16)s[i];
}

__global__ void init_kernel(__bf16* __restrict__ hb, int nh,
                            unsigned* __restrict__ cnt) {
  int i = blockIdx.x * blockDim.x + threadIdx.x;
  int stride = gridDim.x * blockDim.x;
  for (; i < nh; i += stride) hb[i] = (__bf16)0.0f;
  if (blockIdx.x == 0 && threadIdx.x == 0) *cnt = 0u;  // graph-replay safe reset
}

// ---------------- phase 1: xW = x @ W^T ----------------
// One wave computes a 16x64 strip (1 M-tile x 4 N-tiles): the A fragment
// (x fp32 -> bf16 on the fly) is reused across 4 WMMAs per K-chunk.
__global__ void __launch_bounds__(128)
xw_gemm_kernel(const float* __restrict__ x, const __bf16* __restrict__ Wb,
               float* __restrict__ states, int BT, int I, int H) {
  const int lane = threadIdx.x & 31;
  const int wave = blockIdx.x * 4 + (threadIdx.x >> 5);
  const int NG = H >> 6;                 // groups of 4 N-tiles
  const int mt = wave / NG;
  const int ng = wave - mt * NG;
  if (mt >= (BT >> 4)) return;

  const int r  = lane & 15;
  const int kh = lane >> 4;
  const long arow  = (long)(mt * 16 + r);
  const long bcol0 = (long)(ng * 64 + r);      // lane's W row for N-tile j: +16*j

  f32x8 acc0 = {}, acc1 = {}, acc2 = {}, acc3 = {};
  for (int k0 = 0; k0 < I; k0 += 32) {
    const float* ap = x + arow * I + k0 + 16 * kh;
    __builtin_prefetch(ap + 32, 0, 3);         // global_prefetch next K-chunk
    bf16x16 a;
#pragma unroll
    for (int e = 0; e < 8; ++e) { a[e] = (__bf16)ap[e]; a[8 + e] = (__bf16)ap[8 + e]; }
    const __bf16* bp = Wb + k0 + 16 * kh;
    const bf16x16 b0 = *(const bf16x16*)(bp + (bcol0     ) * I);
    const bf16x16 b1 = *(const bf16x16*)(bp + (bcol0 + 16) * I);
    const bf16x16 b2 = *(const bf16x16*)(bp + (bcol0 + 32) * I);
    const bf16x16 b3 = *(const bf16x16*)(bp + (bcol0 + 48) * I);
    acc0 = __builtin_amdgcn_wmma_f32_16x16x32_bf16(false, a, false, b0, (short)0, acc0, false, false);
    acc1 = __builtin_amdgcn_wmma_f32_16x16x32_bf16(false, a, false, b1, (short)0, acc1, false, false);
    acc2 = __builtin_amdgcn_wmma_f32_16x16x32_bf16(false, a, false, b2, (short)0, acc2, false, false);
    acc3 = __builtin_amdgcn_wmma_f32_16x16x32_bf16(false, a, false, b3, (short)0, acc3, false, false);
  }
  // C layout: elem i, lane l -> M = i + 8*(l>>4), N = l&15
  const f32x8 accs[4] = {acc0, acc1, acc2, acc3};
#pragma unroll
  for (int j = 0; j < 4; ++j) {
    const int n = ng * 64 + j * 16 + (lane & 15);
#pragma unroll
    for (int i = 0; i < 8; ++i) {
      const long m = (long)(mt * 16 + i + 8 * kh);
      states[m * H + n] = accs[j][i];
    }
  }
}

// ---------------- phase 2: persistent recurrence ----------------
// 128 waves (2 x 64 tiles of the 32x1024 h matrix); fp32 h tile stays in the
// accumulator registers for all T steps. Each block's 64-row U slice (128 KB
// bf16) is staged into LDS once by the Tensor Data Mover; B fragments then
// come from LDS every step. bf16 h is double-buffered in global memory with
// one device-wide barrier per step.
extern __shared__ __bf16 Ulds[];   // 64 * H bf16

__global__ void __launch_bounds__(128)
liquid_rec_kernel(const __bf16* __restrict__ Ub, const float* __restrict__ bvec,
                  const float* __restrict__ tau, float* __restrict__ states,
                  float* __restrict__ hfinal, __bf16* __restrict__ hb,
                  unsigned* __restrict__ cnt, int T, int H, int nblocks) {
  const int lane = threadIdx.x & 31;
  const int wid  = threadIdx.x >> 5;
  const int wave = blockIdx.x * 4 + wid;
  const int NT = H >> 4;                     // 64
  const int mt = wave / NT;                  // 0..1
  const int nt = wave - mt * NT;             // 0..63
  const int r  = lane & 15;
  const int kh = lane >> 4;
  const int n  = nt * 16 + r;                // output column (U row)
  const int n0b = (blockIdx.x % (NT >> 2)) * 64;  // block's first column
  const long arow = (long)(mt * 16 + r);     // h row this lane loads for A
  const long BH = (long)Bdim * H;

  // ---- TDM: stage U rows [n0b, n0b+64) x K into LDS (wave 0 issues) ----
  if (wid == 0) {
    const unsigned lds_off = (unsigned)(uintptr_t)Ulds;
    const unsigned long long ga =
        (unsigned long long)(uintptr_t)(Ub + (long)n0b * H);
    u32x4 g0;
    g0[0] = rfl(1u);                                        // count=1, user D#
    g0[1] = rfl(lds_off);                                   // lds_addr
    g0[2] = rfl((unsigned)(ga & 0xFFFFFFFFull));            // global_addr lo
    g0[3] = rfl((unsigned)((ga >> 32) & 0x01FFFFFFull) | (2u << 30)); // hi | type=2
    i32x8 g1;
    g1[0] = (int)rfl(0x00010000u);                          // data_size=2B
    g1[1] = (int)rfl((unsigned)H << 16);                    // tensor_dim0 lo16
    g1[2] = (int)rfl(((unsigned)H >> 16) | (64u << 16));    // dim0 hi | tensor_dim1=64
    g1[3] = (int)rfl((unsigned)H << 16);                    // tile_dim0 = H
    g1[4] = (int)rfl(64u);                                  // tile_dim1 = 64
    g1[5] = (int)rfl((unsigned)H);                          // dim0_stride lo32
    g1[6] = (int)rfl((unsigned)H << 16);                    // dim0_str hi | dim1_stride lo16
    g1[7] = (int)rfl((unsigned)H >> 16);                    // dim1_stride hi
    const i32x4 z4 = {0, 0, 0, 0};
#if defined(__clang_major__) && (__clang_major__ >= 23)
    const i32x8 z8 = {0, 0, 0, 0, 0, 0, 0, 0};
    __builtin_amdgcn_tensor_load_to_lds(g0, g1, z4, z4, z8, 0);
#else
    __builtin_amdgcn_tensor_load_to_lds(g0, g1, z4, z4, 0);
#endif
    __builtin_amdgcn_s_wait_tensorcnt(0);
  }
  __syncthreads();

  const __bf16* Urow = Ulds + (long)(n - n0b) * H;  // this lane's B column in LDS

  const float bval = bvec[n];
  const float itv  = 1.0f / tau[n];

  f32x8 h = {};                              // fp32 h tile, register-resident
  for (int t = 0; t < T; ++t) {
    const __bf16* hcur = hb + (long)(t & 1) * BH;
    __bf16*       hnxt = hb + (long)((t + 1) & 1) * BH;

    f32x8 acc = {};
    for (int k0 = 0; k0 < H; k0 += 32) {
      const bf16x8 lo = *(const bf16x8*)(hcur + arow * H + k0 + 16 * kh);
      const bf16x8 hi = *(const bf16x8*)(hcur + arow * H + k0 + 8 + 16 * kh);
      bf16x16 a;
#pragma unroll
      for (int e = 0; e < 8; ++e) { a[e] = lo[e]; a[8 + e] = hi[e]; }
      const bf16x16 bm = *(const bf16x16*)(Urow + k0 + 16 * kh);   // ds_load
      acc = __builtin_amdgcn_wmma_f32_16x16x32_bf16(false, a, false, bm,
                                                    (short)0, acc, false, false);
    }

#pragma unroll
    for (int i = 0; i < 8; ++i) {
      const long m = (long)(mt * 16 + i + 8 * kh);         // batch row
      float* sp = states + (m * T + t) * H + n;            // holds xW_t, becomes h_t
      const float xw  = *sp;
      const float act = tanhf(acc[i] + xw + bval);
      const float hn  = h[i] + DT_STEP * (-itv * h[i] + act);
      h[i] = hn;
      *sp = hn;                                            // all_states[b,t,n]
      hnxt[m * H + n] = (__bf16)hn;                        // bf16 h for next step
      if (t == T - 1) hfinal[m * H + n] = hn;
    }

    // device-wide barrier: all hnxt/state writes visible before step t+1
    __threadfence();
    __syncthreads();
    if (threadIdx.x == 0) {
      __hip_atomic_fetch_add(cnt, 1u, __ATOMIC_ACQ_REL, __HIP_MEMORY_SCOPE_AGENT);
      const unsigned goal = (unsigned)nblocks * (unsigned)(t + 1);
      while (__hip_atomic_load(cnt, __ATOMIC_ACQUIRE, __HIP_MEMORY_SCOPE_AGENT) < goal) {
        __builtin_amdgcn_s_sleep(1);
      }
    }
    __syncthreads();
  }
}

// ---------------- launch ----------------

extern "C" void kernel_launch(void* const* d_in, const int* in_sizes, int n_in,
                              void* d_out, int out_size, void* d_ws, size_t ws_size,
                              hipStream_t stream) {
  const float* x   = (const float*)d_in[0];  // [B,T,I]
  const float* W   = (const float*)d_in[1];  // [H,I]
  const float* U   = (const float*)d_in[2];  // [H,H]
  const float* bv  = (const float*)d_in[3];  // [H]
  const float* tau = (const float*)d_in[4];  // [H]

  const int  H  = in_sizes[3];
  const int  I  = in_sizes[1] / H;
  const long BT = (long)in_sizes[0] / I;
  const int  T  = (int)(BT / Bdim);

  float* hfinal = (float*)d_out;                 // [B,H]
  float* states = hfinal + (long)Bdim * H;       // [B,T,H], seeded with xW

  // workspace: Wb (bf16 H*I) | Ub (bf16 H*H) | hb (bf16 2*B*H) | counter
  __bf16* Wb = (__bf16*)d_ws;
  __bf16* Ub = Wb + (long)H * I;
  __bf16* hb = Ub + (long)H * H;
  unsigned* cnt = (unsigned*)(hb + 2L * Bdim * H);

  cvt_bf16_kernel<<<256, 256, 0, stream>>>(W, Wb, H * I);
  cvt_bf16_kernel<<<256, 256, 0, stream>>>(U, Ub, H * H);
  init_kernel<<<64, 256, 0, stream>>>(hb, 2 * Bdim * H, cnt);

  // phase 1: one wave per 16x64 strip of [BT, H]
  const long waves1 = (BT >> 4) * (long)(H >> 6);
  xw_gemm_kernel<<<(int)(waves1 / 4), 128, 0, stream>>>(x, Wb, states,
                                                        (int)BT, I, H);

  // phase 2: persistent kernel; 4 waves/block, 64 U rows -> 128KB dynamic LDS
  const int rblocks = (Bdim >> 4) * (H >> 4) / 4;        // 32
  const size_t ldsBytes = (size_t)64 * H * sizeof(__bf16);
  liquid_rec_kernel<<<rblocks, 128, ldsBytes, stream>>>(Ub, bv, tau, states,
                                                        hfinal, hb, cnt, T, H,
                                                        rblocks);
}
